// LFIFusion_32220844654886
// MI455X (gfx1250) — matrix operand order repaired and checked by reading, not compiled, not run on hardware
//
#include <hip/hip_runtime.h>
#include <hip/hip_bf16.h>
#include <math.h>
#include <stdint.h>

// ---------------------------------------------------------------------------
// LFI fusion on MI455X (gfx1250).
//   kw_convert : weights fp32 -> bf16 (packed layouts) in workspace
//   k1_in      : Hcat = [Win1;Win2;Win3] @ (gamma_s * out_s)   (bf16 out)
//   k2_trans   : Htra = Wtrans @ Hcat + b_trans                (bf16 out)
//   k3_out     : acc_j = Wout_j @ Htra[jD:(j+1)D] (3 GEMMs per tile),
//                then sigmoid -> softmax over j -> fused output (fp32)
// All GEMMs: v_wmma_f32_16x16x32_bf16; B operand double-buffered into LDS via
// GLOBAL_LOAD_ASYNC_TO_LDS_B128 (+ s_wait_asynccnt), one shared copy per WG.
// k2 uses a square 128x128 block tile (4Mx2N waves, 32x64 per wave) to
// balance A/B L2 streams on the dominant GEMM.
// ---------------------------------------------------------------------------

typedef __attribute__((ext_vector_type(16))) __bf16 v16bf;
typedef __attribute__((ext_vector_type(8)))  float  v8f;
typedef __attribute__((ext_vector_type(4)))  float  v4f;

#define D_     512
#define TD_    1536          // 3*D
#define HW_    1024          // 32*32
#define BATCH_ 32
#define NPIX_  (BATCH_ * HW_)  // 32768

// LDS staging geometry (N tile = 128 columns).
#define KC1        32                      // K-chunk for fp32 staging (k1)
#define F32_ROWB   512                     // 128 cols * 4B
#define F32_STRIDE 528                     // +16B pad -> lane bank stride 4
#define BUF1       (KC1 * F32_STRIDE)      // 16896 B per buffer

#define KC2        64                      // K-chunk for bf16 staging (k2/k3)
#define BF_ROWB    256                     // 128 cols * 2B
#define BF_STRIDE  272                     // +16B pad -> lane bank stride 4
#define BUF2       (KC2 * BF_STRIDE)       // 17408 B per buffer

// ---- WMMA wrapper ----------------------------------------------------------
__device__ __forceinline__ v8f wmma_bf16f32(v16bf a, v16bf b, v8f c) {
    return __builtin_amdgcn_wmma_f32_16x16x32_bf16(false, a, false, b,
                                                   (short)0, c, false, false);
}

// ---- Async global->LDS tile copy (cooperative, 256 threads) ----------------
template <int ROWS, int ROWB, int STRIDE>
__device__ __forceinline__ void issue_tile(const char* g0, size_t gstride,
                                           uint32_t lds_base) {
    constexpr int PERROW = ROWB / 16;
    constexpr int XPT = (ROWS * PERROW) / 256;
    static_assert(XPT == 4, "expect 4 async issues per wave per chunk");
    const int t = threadIdx.x;
#pragma unroll
    for (int i = 0; i < XPT; ++i) {
        const int linear = t + 256 * i;
        const int row = linear / PERROW;
        const int c   = linear % PERROW;
        const uint64_t ga = (uint64_t)(uintptr_t)(g0 + (size_t)row * gstride + c * 16);
        const uint32_t la = lds_base + (uint32_t)(row * STRIDE + c * 16);
        asm volatile("global_load_async_to_lds_b128 %0, %1, off"
                     :: "v"(la), "v"(ga) : "memory");
    }
}

// ---- A fragment: 16x32 bf16 weight tile, row-major W[ldw] ------------------
__device__ __forceinline__ v16bf load_a_frag(const __bf16* __restrict__ W,
                                             int ldw, int m0, int k0) {
    const int lane = threadIdx.x & 31;
    const int half = lane >> 4;
    const int r    = lane & 15;
    const __bf16* p = W + (size_t)(m0 + r) * ldw + (k0 + half * 8);
    union { v16bf v; v4f f[2]; } u;
    u.f[0] = *reinterpret_cast<const v4f*>(p);
    u.f[1] = *reinterpret_cast<const v4f*>(p + 16);
    return u.v;
}

// ---- B fragments from the LDS-staged tile (lane = N column) ----------------
__device__ __forceinline__ v16bf lds_b_frag_bf16(const char* buf, int kk, int nl) {
    const int lane = threadIdx.x & 31;
    const int half = lane >> 4;
    const char* p = buf + kk * BF_STRIDE + (nl + (lane & 15)) * 2;
    v16bf v;
#pragma unroll
    for (int i = 0; i < 16; ++i) {
        const int k = (i < 8) ? (half * 8 + i) : (16 + half * 8 + (i - 8));
        v[i] = *reinterpret_cast<const __bf16*>(p + k * BF_STRIDE);
    }
    return v;
}

__device__ __forceinline__ v16bf lds_b_frag_f32(const char* buf, int kk, int nl,
                                                float scale) {
    const int lane = threadIdx.x & 31;
    const int half = lane >> 4;
    const char* p = buf + kk * F32_STRIDE + (nl + (lane & 15)) * 4;
    v16bf v;
#pragma unroll
    for (int i = 0; i < 16; ++i) {
        const int k = (i < 8) ? (half * 8 + i) : (16 + half * 8 + (i - 8));
        v[i] = (__bf16)(*reinterpret_cast<const float*>(p + k * F32_STRIDE) * scale);
    }
    return v;
}

// ---- C/D epilogue: lane<16 -> col n0+lane rows m0+r; lane>=16 -> rows m0+8+r
template <int NF>
__device__ __forceinline__ void store_tile_bf16(__bf16* __restrict__ Yb, int ldy,
                                                int m0, int n0,
                                                const v8f (&acc)[2][NF],
                                                const float* __restrict__ bias,
                                                int bias_mask) {
    const int lane = threadIdx.x & 31;
    const int half = lane >> 4;
    const int col  = lane & 15;
#pragma unroll
    for (int mi = 0; mi < 2; ++mi)
#pragma unroll
        for (int ni = 0; ni < NF; ++ni) {
            const int c = n0 + ni * 16 + col;
#pragma unroll
            for (int r = 0; r < 8; ++r) {
                const int row = m0 + mi * 16 + half * 8 + r;
                const float val = acc[mi][ni][r] + bias[row & bias_mask];
                Yb[(size_t)row * ldy + c] = (__bf16)val;
            }
        }
}

// ---------------------------------------------------------------------------
// kw_convert: fp32 weights -> bf16 workspace.
// ---------------------------------------------------------------------------
__global__ void kw_convert(const float* __restrict__ Wi1, const float* __restrict__ Wi2,
                           const float* __restrict__ Wi3, const float* __restrict__ Wtr,
                           const float* __restrict__ Wo1, const float* __restrict__ Wo2,
                           const float* __restrict__ Wo3,
                           __bf16* __restrict__ Wm1, __bf16* __restrict__ Wt,
                           __bf16* __restrict__ Wo) {
    const int NW1 = TD_ * D_;
    const int NWT = TD_ * TD_;
    const int NWO = 3 * D_ * D_;
    const int total = NW1 + NWT + NWO;
    for (int idx = blockIdx.x * blockDim.x + threadIdx.x; idx < total;
         idx += gridDim.x * blockDim.x) {
        if (idx < NW1) {
            const int row = idx >> 9, k = idx & 511, s = row >> 9;
            const float* src = (s == 0) ? Wi1 : (s == 1) ? Wi2 : Wi3;
            Wm1[idx] = (__bf16)src[((row & 511) << 9) | k];
        } else if (idx < NW1 + NWT) {
            const int j = idx - NW1;
            Wt[j] = (__bf16)Wtr[j];
        } else {
            const int j = idx - NW1 - NWT;
            const int s = j >> 18;
            const float* src = (s == 0) ? Wo1 : (s == 1) ? Wo2 : Wo3;
            Wo[j] = (__bf16)src[j & 262143];
        }
    }
}

// ---------------------------------------------------------------------------
// k1_in: Hcat[b][row][hw] = Wm1[row][:] . (gamma_s * out_s[b][:][hw]) + b_in_s
// grid = (NPIX/128, TD/64); block = 256 (8 waves = 2(M) x 4(N), 32x32 each)
// ---------------------------------------------------------------------------
__global__ __launch_bounds__(256) void k1_in(
    const float* __restrict__ x0, const float* __restrict__ x1, const float* __restrict__ x2,
    const float* __restrict__ g0, const float* __restrict__ g1, const float* __restrict__ g2,
    const float* __restrict__ b1, const float* __restrict__ b2, const float* __restrict__ b3,
    const __bf16* __restrict__ Wm1, __bf16* __restrict__ Hcat) {
    __shared__ __align__(16) char smem[2 * BUF1];
    const int wave = threadIdx.x >> 5;
    const int m0   = blockIdx.y * 64 + (wave & 1) * 32;
    const int nblk = blockIdx.x * 128;
    const int nl   = (wave >> 1) * 32;
    const int s    = m0 >> 9;
    const float* X    = (s == 0) ? x0 : (s == 1) ? x1 : x2;
    const float* bias = (s == 0) ? b1 : (s == 1) ? b2 : b3;
    const float scale = (s == 0) ? g0[0] : (s == 1) ? g1[0] : g2[0];
    const int b  = nblk >> 10;
    const int hw = nblk & 1023;
    const float* Xb = X + (size_t)b * (D_ * HW_) + hw;
    const uint32_t lds0 = (uint32_t)(uintptr_t)&smem[0];

    issue_tile<KC1, F32_ROWB, F32_STRIDE>((const char*)Xb, sizeof(float) * HW_, lds0);
    v8f acc[2][2] = {};
    const int NCH = D_ / KC1;  // 16
    for (int g = 0; g < NCH; ++g) {
        if (g + 1 < NCH) {
            issue_tile<KC1, F32_ROWB, F32_STRIDE>(
                (const char*)(Xb + (size_t)(g + 1) * KC1 * HW_),
                sizeof(float) * HW_, lds0 + ((g + 1) & 1) * BUF1);
            asm volatile("s_wait_asynccnt 0x4" ::: "memory");
        } else {
            asm volatile("s_wait_asynccnt 0x0" ::: "memory");
        }
        __syncthreads();
        const char* buf = &smem[(g & 1) * BUF1];
        const int k0 = g * KC1;
        const v16bf a0  = load_a_frag(Wm1, D_, m0, k0);
        const v16bf a1  = load_a_frag(Wm1, D_, m0 + 16, k0);
        const v16bf bb0 = lds_b_frag_f32(buf, 0, nl, scale);
        const v16bf bb1 = lds_b_frag_f32(buf, 0, nl + 16, scale);
        acc[0][0] = wmma_bf16f32(a0, bb0, acc[0][0]);
        acc[0][1] = wmma_bf16f32(a0, bb1, acc[0][1]);
        acc[1][0] = wmma_bf16f32(a1, bb0, acc[1][0]);
        acc[1][1] = wmma_bf16f32(a1, bb1, acc[1][1]);
        __syncthreads();
    }
    __bf16* Yb = Hcat + (size_t)b * (TD_ * HW_);
    store_tile_bf16<2>(Yb, HW_, m0, hw + nl, acc, bias, 511);
}

// ---------------------------------------------------------------------------
// k2_trans: Htra = Wtrans @ Hcat + b_trans   (K = 1536, bf16 in/out)
// Square 128x128 block tile: 8 waves = 4(M) x 2(N), wave tile 32x64.
// grid = (NPIX/128, TD/128)
// ---------------------------------------------------------------------------
__global__ __launch_bounds__(256) void k2_trans(
    const __bf16* __restrict__ Wt, const float* __restrict__ btr,
    const __bf16* __restrict__ Hcat, __bf16* __restrict__ Htra) {
    __shared__ __align__(16) char smem[2 * BUF2];
    const int wave = threadIdx.x >> 5;
    const int m0   = blockIdx.y * 128 + (wave & 3) * 32;
    const int nblk = blockIdx.x * 128;
    const int nl   = (wave >> 2) * 64;
    const int b  = nblk >> 10;
    const int hw = nblk & 1023;
    const __bf16* Xb = Hcat + (size_t)b * (TD_ * HW_) + hw;
    const uint32_t lds0 = (uint32_t)(uintptr_t)&smem[0];

    issue_tile<KC2, BF_ROWB, BF_STRIDE>((const char*)Xb, 2 * HW_, lds0);
    v8f acc[2][4] = {};
    const int NCH = TD_ / KC2;  // 24
    for (int g = 0; g < NCH; ++g) {
        if (g + 1 < NCH) {
            issue_tile<KC2, BF_ROWB, BF_STRIDE>(
                (const char*)(Xb + (size_t)(g + 1) * KC2 * HW_),
                2 * HW_, lds0 + ((g + 1) & 1) * BUF2);
            asm volatile("s_wait_asynccnt 0x4" ::: "memory");
        } else {
            asm volatile("s_wait_asynccnt 0x0" ::: "memory");
        }
        __syncthreads();
        const char* buf = &smem[(g & 1) * BUF2];
#pragma unroll
        for (int kk = 0; kk < KC2; kk += 32) {
            const int k0 = g * KC2 + kk;
            const v16bf a0 = load_a_frag(Wt, TD_, m0, k0);
            const v16bf a1 = load_a_frag(Wt, TD_, m0 + 16, k0);
#pragma unroll
            for (int ni = 0; ni < 4; ++ni) {
                const v16bf bb = lds_b_frag_bf16(buf, kk, nl + ni * 16);
                acc[0][ni] = wmma_bf16f32(a0, bb, acc[0][ni]);
                acc[1][ni] = wmma_bf16f32(a1, bb, acc[1][ni]);
            }
        }
        __syncthreads();
    }
    __bf16* Yb = Htra + (size_t)b * (TD_ * HW_);
    store_tile_bf16<4>(Yb, HW_, m0, hw + nl, acc, btr, TD_ - 1);
}

// ---------------------------------------------------------------------------
// k3_out: three GEMMs (one per gate) + sigmoid + softmax + gamma-weighted sum.
// grid = (NPIX/128, D/64); 8 waves = 2(M) x 4(N), 32x32 each.
// ---------------------------------------------------------------------------
__global__ __launch_bounds__(256) void k3_out(
    const __bf16* __restrict__ Wo, const __bf16* __restrict__ Htra,
    const float* __restrict__ bo0, const float* __restrict__ bo1, const float* __restrict__ bo2,
    const float* __restrict__ x0, const float* __restrict__ x1, const float* __restrict__ x2,
    const float* __restrict__ g0, const float* __restrict__ g1, const float* __restrict__ g2,
    float* __restrict__ Y) {
    __shared__ __align__(16) char smem[2 * BUF2];
    const int wave = threadIdx.x >> 5;
    const int lane = threadIdx.x & 31;
    const int m0   = blockIdx.y * 64 + (wave & 1) * 32;
    const int nblk = blockIdx.x * 128;
    const int nl   = (wave >> 1) * 32;
    const int b  = nblk >> 10;
    const int hw = nblk & 1023;
    const __bf16* Xb = Htra + (size_t)b * (TD_ * HW_) + hw;
    const uint32_t lds0 = (uint32_t)(uintptr_t)&smem[0];
    const int NCH = TD_ / KC2;  // 24, 8 per gate j

    issue_tile<KC2, BF_ROWB, BF_STRIDE>((const char*)Xb, 2 * HW_, lds0);
    v8f acc[3][2][2] = {};
#pragma unroll
    for (int j = 0; j < 3; ++j) {
        const __bf16* Wj = Wo + (size_t)j * (D_ * D_);
        for (int c2 = 0; c2 < 8; ++c2) {
            const int g = j * 8 + c2;
            if (g + 1 < NCH) {
                issue_tile<KC2, BF_ROWB, BF_STRIDE>(
                    (const char*)(Xb + (size_t)(g + 1) * KC2 * HW_),
                    2 * HW_, lds0 + ((g + 1) & 1) * BUF2);
                asm volatile("s_wait_asynccnt 0x4" ::: "memory");
            } else {
                asm volatile("s_wait_asynccnt 0x0" ::: "memory");
            }
            __syncthreads();
            const char* buf = &smem[(g & 1) * BUF2];
#pragma unroll
            for (int kk = 0; kk < KC2; kk += 32) {
                const int kl = c2 * KC2 + kk;   // column within Wj (0..511)
                const v16bf a0  = load_a_frag(Wj, D_, m0, kl);
                const v16bf a1  = load_a_frag(Wj, D_, m0 + 16, kl);
                const v16bf bb0 = lds_b_frag_bf16(buf, kk, nl);
                const v16bf bb1 = lds_b_frag_bf16(buf, kk, nl + 16);
                acc[j][0][0] = wmma_bf16f32(a0, bb0, acc[j][0][0]);
                acc[j][0][1] = wmma_bf16f32(a0, bb1, acc[j][0][1]);
                acc[j][1][0] = wmma_bf16f32(a1, bb0, acc[j][1][0]);
                acc[j][1][1] = wmma_bf16f32(a1, bb1, acc[j][1][1]);
            }
            __syncthreads();
        }
    }

    const float G0 = g0[0], G1 = g1[0], G2 = g2[0];
    const int half = lane >> 4, col = lane & 15;
    const size_t base = (size_t)b * (D_ * HW_);
#pragma unroll
    for (int mi = 0; mi < 2; ++mi)
#pragma unroll
        for (int ni = 0; ni < 2; ++ni) {
            const int c = hw + nl + ni * 16 + col;
#pragma unroll
            for (int r = 0; r < 8; ++r) {
                const int row = m0 + mi * 16 + half * 8 + r;
                const float a0 = acc[0][mi][ni][r] + bo0[row];
                const float a1 = acc[1][mi][ni][r] + bo1[row];
                const float a2 = acc[2][mi][ni][r] + bo2[row];
                const float s0 = 1.f / (1.f + __expf(-a0));
                const float s1 = 1.f / (1.f + __expf(-a1));
                const float s2 = 1.f / (1.f + __expf(-a2));
                const float mx = fmaxf(s0, fmaxf(s1, s2));
                const float e0 = __expf(s0 - mx);
                const float e1 = __expf(s1 - mx);
                const float e2 = __expf(s2 - mx);
                const float inv = 1.f / (e0 + e1 + e2);
                const size_t idx = base + (size_t)row * HW_ + c;
                Y[idx] = (e0 * G0 * x0[idx] + e1 * G1 * x1[idx] + e2 * G2 * x2[idx]) * inv;
            }
        }
}

// ---------------------------------------------------------------------------
// Workspace layout (bytes). Total ~199.5 MiB.
// ---------------------------------------------------------------------------
static constexpr size_t SZ_WM1  = (size_t)TD_ * D_ * 2;
static constexpr size_t SZ_WT   = (size_t)TD_ * TD_ * 2;
static constexpr size_t SZ_WO   = (size_t)3 * D_ * D_ * 2;
static constexpr size_t SZ_HCAT = (size_t)BATCH_ * TD_ * HW_ * 2;
static constexpr size_t OFF_WM1  = 0;
static constexpr size_t OFF_WT   = OFF_WM1 + SZ_WM1;
static constexpr size_t OFF_WO   = OFF_WT + SZ_WT;
static constexpr size_t OFF_HCAT = OFF_WO + SZ_WO;
static constexpr size_t OFF_HTRA = OFF_HCAT + SZ_HCAT;

extern "C" void kernel_launch(void* const* d_in, const int* in_sizes, int n_in,
                              void* d_out, int out_size, void* d_ws, size_t ws_size,
                              hipStream_t stream) {
    (void)in_sizes; (void)n_in; (void)out_size; (void)ws_size;
    const float* out0   = (const float*)d_in[0];
    const float* out1   = (const float*)d_in[1];
    const float* out2   = (const float*)d_in[2];
    const float* gamma0 = (const float*)d_in[3];
    const float* gamma1 = (const float*)d_in[4];
    const float* gamma2 = (const float*)d_in[5];
    const float* W_in1  = (const float*)d_in[6];
    const float* b_in1  = (const float*)d_in[7];
    const float* W_in2  = (const float*)d_in[8];
    const float* b_in2  = (const float*)d_in[9];
    const float* W_in3  = (const float*)d_in[10];
    const float* b_in3  = (const float*)d_in[11];
    const float* W_tr   = (const float*)d_in[12];
    const float* b_tr   = (const float*)d_in[13];
    const float* W_out1 = (const float*)d_in[14];
    const float* b_out1 = (const float*)d_in[15];
    const float* W_out2 = (const float*)d_in[16];
    const float* b_out2 = (const float*)d_in[17];
    const float* W_out3 = (const float*)d_in[18];
    const float* b_out3 = (const float*)d_in[19];
    float* Y = (float*)d_out;

    char* ws = (char*)d_ws;
    __bf16* Wm1  = (__bf16*)(ws + OFF_WM1);
    __bf16* Wt   = (__bf16*)(ws + OFF_WT);
    __bf16* Wo   = (__bf16*)(ws + OFF_WO);
    __bf16* Hcat = (__bf16*)(ws + OFF_HCAT);
    __bf16* Htra = (__bf16*)(ws + OFF_HTRA);

    kw_convert<<<dim3(2048), dim3(256), 0, stream>>>(
        W_in1, W_in2, W_in3, W_tr, W_out1, W_out2, W_out3, Wm1, Wt, Wo);

    k1_in<<<dim3(NPIX_ / 128, TD_ / 64), dim3(256), 0, stream>>>(
        out0, out1, out2, gamma0, gamma1, gamma2, b_in1, b_in2, b_in3, Wm1, Hcat);

    k2_trans<<<dim3(NPIX_ / 128, TD_ / 128), dim3(256), 0, stream>>>(
        Wt, b_tr, Hcat, Htra);

    k3_out<<<dim3(NPIX_ / 128, D_ / 64), dim3(256), 0, stream>>>(
        Wo, Htra, b_out1, b_out2, b_out3, out0, out1, out2,
        gamma0, gamma1, gamma2, Y);
}
